// MinimumErrorRateLoss_26259430047961
// MI455X (gfx1250) — compile-verified, matrix operand order checked by reading, not run on hardware
//
#include <hip/hip_runtime.h>
#include <hip/hip_bf16.h>
#include <stdint.h>

// Problem constants (from setup_inputs)
#define N_ 128
#define M_ 32
#define R_ 256
#define H_ 288
#define B_ (N_ * M_)

typedef __attribute__((ext_vector_type(16))) _Float16 v16h;
typedef __attribute__((ext_vector_type(8)))  float    v8f;
typedef __attribute__((ext_vector_type(4)))  unsigned int v4u;
typedef __attribute__((ext_vector_type(8)))  int      v8i;
typedef __attribute__((ext_vector_type(4)))  int      v4i;

static __device__ __forceinline__ int imin_(int a, int b) { return a < b ? a : b; }

// ---------------------------------------------------------------------------
// Kernel 1: batched Levenshtein via min-plus scan, one WAVE per batch element.
// State kept in m-space: v[i] = E[i] - i, 8 entries per lane in VGPRs.
// Per hyp step t (state E_t -> E_{t+1}):
//   m[i]   = min(v[i] + 1, v[i-1] - eq[i])      (elementwise; eq = match)
//   v_new  = prefix-min(m) with boundary m[0] = t+1
// so the deletion pass IS the new row -- no index add/sub needed.
// Scan = 7-min local prefix + 5-step wave32 shuffle scan. No LDS in the loop.
// Grid: 512 blocks x 256 threads; block (n, mgroup) covers waves m = mg*8+w.
// Ref column staged once per block via the Tensor Data Mover when available.
// ---------------------------------------------------------------------------
__global__ __launch_bounds__(256) void mer_editdist_kernel(
    const int* __restrict__ ref,   // (R, N) int32 token ids
    const int* __restrict__ hyp,   // (H, N, M) int32 token ids
    float* __restrict__ er)        // (N, M) float: dist / R
{
    __shared__ unsigned int ref_s[R_];

    const int n     = blockIdx.x >> 2;
    const int mbase = (blockIdx.x & 3) * 8;
    const int w     = threadIdx.x >> 5;   // wave id within block (0..7)
    const int lane  = threadIdx.x & 31;

    // ---- Stage ref[:, n] (element stride N_) into LDS ----------------------
#if __has_builtin(__builtin_amdgcn_tensor_load_to_lds) && __has_builtin(__builtin_amdgcn_s_wait_tensorcnt)
    if (w == 0) {
        // D# per CDNA5 ISA §8: 2D u32 tensor, tile 1 x R_, dim0_stride = N_
        unsigned long long ga = (unsigned long long)(uintptr_t)(ref + n);
        unsigned int lds_off  = (unsigned int)(uintptr_t)&ref_s[0];
        v4u g0 = { 1u, lds_off,
                   (unsigned int)(ga & 0xffffffffu),
                   (unsigned int)((ga >> 32) & 0x1ffffffu) | (2u << 30) };
        v8i g1 = { (int)(2u << 16),                        // data_size=4B
                   (int)(((unsigned)N_) << 16),            // tensor_dim0
                   (int)(((unsigned)R_ & 0xffffu) << 16),  // tensor_dim1 lo16
                   (int)(1u << 16),                        // tile_dim0 = 1
                   (int)(R_),                              // tile_dim1 = R_
                   (int)(N_),                              // dim0_stride
                   0, 0 };
        v4i gz = { 0, 0, 0, 0 };
#if defined(__clang_major__) && (__clang_major__ >= 23)
        v8i gz8 = { 0, 0, 0, 0, 0, 0, 0, 0 };
        __builtin_amdgcn_tensor_load_to_lds(g0, g1, gz, gz, gz8, 0);
#else
        __builtin_amdgcn_tensor_load_to_lds(g0, g1, gz, gz, 0);
#endif
        __builtin_amdgcn_s_wait_tensorcnt(0);
    }
#else
    for (int i = threadIdx.x; i < R_; i += 256)
        ref_s[i] = (unsigned int)ref[(size_t)i * N_ + n];
#endif
    __syncthreads();

    // Register-resident ref chunk + row (m-space): entry e <-> i = lane*8+e+1
    int rf[8], v[8];
#pragma unroll
    for (int e = 0; e < 8; ++e) {
        rf[e] = (int)ref_s[lane * 8 + e];
        v[e]  = 0;                       // E_0[i] = i  ->  v = 0
    }

    const int b   = n * M_ + mbase + w;                 // wave-uniform
    const int b_u = __builtin_amdgcn_readfirstlane(b);  // force SGPR base
    const int* hcol = hyp + b_u;

    for (int t = 0; t < H_; ++t) {
        const int h = hcol[(size_t)t * B_];             // scalar load per step
        if ((t & 7) == 0 && t + 8 < H_)
            __builtin_prefetch(hcol + (size_t)(t + 8) * B_, 0, 0);

        // elementwise pass in m-space: m[i] = min(v[i]+1, v[i-1]-eq)
        int prev = __shfl_up(v[7], 1, 32);   // v_old[i-1] for e=0
        if (lane == 0) prev = t;             // boundary v_old[0] = E_t[0]-0 = t
        int m[8];
#pragma unroll
        for (int e = 0; e < 8; ++e) {
            const int sub = (rf[e] == h) ? (prev - 1) : prev;  // v[i-1]-eq
            m[e] = imin_(v[e] + 1, sub);
            prev = v[e];
        }
        // local inclusive prefix-min over this lane's 8 entries
#pragma unroll
        for (int e = 1; e < 8; ++e) m[e] = imin_(m[e], m[e - 1]);
        // wave32 inclusive min-scan of lane totals
        int I = m[7];
#pragma unroll
        for (int d = 1; d < 32; d <<= 1) {
            const int s = __shfl_up(I, d, 32);
            if (lane >= d) I = imin_(I, s);
        }
        const int X      = __shfl_up(I, 1, 32);            // exclusive scan
        const int before = (lane == 0) ? (t + 1)           // boundary m[0]=t+1
                                       : imin_(X, t + 1);
#pragma unroll
        for (int e = 0; e < 8; ++e)
            v[e] = imin_(before, m[e]);                    // v_new = prefix-min
    }

    if (lane == 31)  // entry i = R_ lives in lane 31, e = 7: dist = v + R_
        er[b_u] = (float)(v[7] + R_) * (1.0f / (float)R_);
}

// ---------------------------------------------------------------------------
// Kernel 2: softmax + WMMA reduction (one wave, EXEC all ones).
// Per 16-row group: D = Er(16x32,f16) x P^T(32x16,f16) via
// v_wmma_f32_16x16x32_f16; diag(D) = dot(er_n, p_n).
// loss = (1/NM) * sum_n (dot_n - mean(er_n)).
// ---------------------------------------------------------------------------
__global__ __launch_bounds__(32) void mer_reduce_kernel(
    const float* __restrict__ lp,
    const float* __restrict__ er,
    float* __restrict__ out)
{
    __shared__ _Float16 er16[N_ * M_];
    __shared__ _Float16 p16[N_ * M_];
    __shared__ float    dmat[32 * 8];
    __shared__ float    redbuf[32];

    const int lane = threadIdx.x;

    float meanacc = 0.0f;
    for (int k = 0; k < 4; ++k) {
        const int r = lane * 4 + k;
        float mx = -1e30f;
        for (int m = 0; m < M_; ++m) mx = fmaxf(mx, lp[r * M_ + m]);
        float s = 0.0f;
        for (int m = 0; m < M_; ++m) s += __expf(lp[r * M_ + m] - mx);
        const float inv = 1.0f / s;
        float es = 0.0f;
        for (int m = 0; m < M_; ++m) {
            p16[r * M_ + m]  = (_Float16)(__expf(lp[r * M_ + m] - mx) * inv);
            const float ev   = er[r * M_ + m];
            er16[r * M_ + m] = (_Float16)ev;
            es += ev;
        }
        meanacc += es * (1.0f / (float)M_);
    }
    __syncthreads();

    // Fragment assembly per ISA 16-bit A layout:
    // lanes 0-15 hold K={0..7,16..23}, lanes 16-31 hold K={8..15,24..31}.
    float dotacc = 0.0f;
    const int half = (lane >> 4) & 1;
    const int mloc = lane & 15;
    for (int g = 0; g < 8; ++g) {
        union { v16h v; _Float16 e[16]; } a, bfr;
        const int mrow = g * 16 + mloc;
#pragma unroll
        for (int j = 0; j < 8; ++j) {
            const int k0 = (j < 4) ? (2 * j + half * 8)
                                   : (16 + 2 * (j - 4) + half * 8);
            a.e[2 * j]       = er16[mrow * M_ + k0];   // A[m][k]
            a.e[2 * j + 1]   = er16[mrow * M_ + k0 + 1];
            bfr.e[2 * j]     = p16[mrow * M_ + k0];    // B[k][n] = P^T
            bfr.e[2 * j + 1] = p16[mrow * M_ + k0 + 1];
        }
        v8f c = {};
        c = __builtin_amdgcn_wmma_f32_16x16x32_f16(
                false, a.v, false, bfr.v, (short)0, c, false, false);
        __syncthreads();
#pragma unroll
        for (int j = 0; j < 8; ++j) dmat[lane * 8 + j] = c[j];
        __syncthreads();
        // diag r: r<8 -> lane r VGPR r; r>=8 -> lane r+16 VGPR r-8
        if (lane < 16) {
            const int src = (lane < 8) ? lane : (lane + 16);
            dotacc += dmat[src * 8 + (lane & 7)];
        }
    }

    redbuf[lane] = dotacc - meanacc;
    __syncthreads();
    if (lane == 0) {
        float s = 0.0f;
        for (int i = 0; i < 32; ++i) s += redbuf[i];
        out[0] = s * (1.0f / (float)(N_ * M_));
    }
}

extern "C" void kernel_launch(void* const* d_in, const int* in_sizes, int n_in,
                              void* d_out, int out_size, void* d_ws, size_t ws_size,
                              hipStream_t stream) {
    (void)in_sizes; (void)n_in; (void)out_size; (void)ws_size;
    const float* lp  = (const float*)d_in[0];
    const int*   ref = (const int*)d_in[1];
    const int*   hyp = (const int*)d_in[2];
    float* er_ws = (float*)d_ws;
    float* out   = (float*)d_out;

    mer_editdist_kernel<<<dim3(N_ * 4), dim3(256), 0, stream>>>(ref, hyp, er_ws);
    mer_reduce_kernel<<<dim3(1), dim3(32), 0, stream>>>(lp, er_ws, out);
}
